// PKVAttention_19670950216256
// MI455X (gfx1250) — matrix operand order repaired and checked by reading, not compiled, not run on hardware
//
#include <hip/hip_runtime.h>
#include <hip/hip_bf16.h>
#include <stdint.h>

// Problem constants (reference: B=2, S=2048, D=2048, H=16, HD=128, RD=64)
#define BB 2
#define SS 2048
#define DD 2048
#define HH 16

typedef __attribute__((ext_vector_type(16))) __bf16 v16bf;
typedef __attribute__((ext_vector_type(8)))  float  v8f;

union ABfrag { v16bf v; uint32_t u[8]; };
union Q16    { uint4 q; uint16_t h[8]; };

__device__ __forceinline__ uint16_t f2bf(float f) {
  uint32_t u = __float_as_uint(f);
  uint32_t r = u + 0x7FFFu + ((u >> 16) & 1u);   // round-to-nearest-even
  return (uint16_t)(r >> 16);
}
__device__ __forceinline__ float bf2f(uint16_t h) {
  return __uint_as_float(((uint32_t)h) << 16);
}
__device__ __forceinline__ v8f vzero() {
  v8f z;
#pragma unroll
  for (int i = 0; i < 8; ++i) z[i] = 0.0f;
  return z;
}

// CDNA5 async copy: 16B per lane, global -> LDS, tracked by ASYNCcnt.
// Low 32 bits of a generic LDS pointer are the LDS byte address (ISA 10.2).
__device__ __forceinline__ void async_copy_b128(const void* gsrc, void* ldst) {
  uint32_t l = (uint32_t)(uintptr_t)ldst;
  asm volatile("global_load_async_to_lds_b128 %0, %1, off"
               :: "v"(l), "v"(gsrc) : "memory");
}
__device__ __forceinline__ void wait_async() {
  asm volatile("s_wait_asynccnt 0" ::: "memory");
}

// ---------------------------------------------------------------------------
// fp32 -> bf16 conversion
// ---------------------------------------------------------------------------
__global__ void f32_to_bf16_kernel(const float* __restrict__ in,
                                   uint16_t* __restrict__ out, int n) {
  int t = blockIdx.x * blockDim.x + threadIdx.x;
  if (t < n) out[t] = f2bf(in[t]);
}

// ---------------------------------------------------------------------------
// Tiled bf16 WMMA GEMM: C[M,N] = A[M,K] * B[K,N], A/B row-major bf16.
// Block 256 threads = 8 waves (2x4), tile 128x128, k-step 32.
// A tiles: async DMA to LDS. B tiles: global->reg fetch at loop top,
// reg->LDS (transposed) commit after compute, so the load latency hides
// behind the WMMA work of the current k-step.
// ---------------------------------------------------------------------------
__device__ __forceinline__ void storeC(float* p, float v)    { *p = v; }
__device__ __forceinline__ void storeC(uint16_t* p, float v) { *p = f2bf(v); }

template <typename OUT>
__global__ __launch_bounds__(256) void gemm_bf16(const uint16_t* __restrict__ A,
                                                 const uint16_t* __restrict__ B,
                                                 OUT* __restrict__ C,
                                                 int M, int N, int K) {
  constexpr int BM = 128, BN = 128, BK = 32;
  constexpr int LDA = 40;   // 80B rows -> every 16B chunk stays 16B aligned
  constexpr int LDBT = 34;  // transposed B tile, 4B-aligned dword pairs
  __shared__ uint16_t sA[2][BM][LDA];
  __shared__ uint16_t sBt[2][BN][LDBT];

  const int tid  = threadIdx.x;
  const int lane = tid & 31, wid = tid >> 5;
  const int hi = lane >> 4, lr = lane & 15;
  const int m0 = blockIdx.y * BM, n0 = blockIdx.x * BN;
  const int wm = (wid >> 2) * 64, wn = (wid & 3) * 32;
  const int nkt = K / BK;

  auto loadA = [&](int buf, int k0) {
#pragma unroll
    for (int j = 0; j < 2; ++j) {
      int c = tid + j * 256;             // 512 chunks of 16B (128 rows x 4)
      int row = c >> 2, cc = (c & 3) * 8;
      async_copy_b128(A + (size_t)(m0 + row) * K + k0 + cc, &sA[buf][row][cc]);
    }
  };
  Q16 breg[2];
  auto fetchB = [&](int k0) {
#pragma unroll
    for (int j = 0; j < 2; ++j) {
      int c = tid + j * 256;             // 512 chunks: 32 k-rows x 16
      int kr = c >> 4, nc = (c & 15) * 8;
      breg[j].q = *(const uint4*)(B + (size_t)(k0 + kr) * N + n0 + nc);
    }
  };
  auto commitB = [&](int buf) {
#pragma unroll
    for (int j = 0; j < 2; ++j) {
      int c = tid + j * 256;
      int kr = c >> 4, nc = (c & 15) * 8;
#pragma unroll
      for (int e = 0; e < 8; ++e) sBt[buf][nc + e][kr] = breg[j].h[e];
    }
  };

  v8f acc[4][2];
#pragma unroll
  for (int im = 0; im < 4; ++im)
#pragma unroll
    for (int jn = 0; jn < 2; ++jn) acc[im][jn] = vzero();

  fetchB(0);
  loadA(0, 0);
  commitB(0);
  wait_async();
  __syncthreads();

  for (int kt = 0; kt < nkt; ++kt) {
    const int buf = kt & 1;
    const bool more = (kt + 1) < nkt;
    if (more) {
      loadA(buf ^ 1, (kt + 1) * BK);     // async DMA overlaps this k-step
      fetchB((kt + 1) * BK);             // global->reg overlaps this k-step
      if (kt + 2 < nkt) {                // warm L2/WGP$ two steps ahead
        __builtin_prefetch(A + (size_t)(m0 + (tid >> 2)) * K + (kt + 2) * BK, 0, 0);
        __builtin_prefetch(B + (size_t)((kt + 2) * BK + (tid >> 4)) * N + n0, 0, 0);
      }
    }

    // A fragments: 16x32 bf16, lane = row, K pairs 2i/2i+1 (+16 hi)
    ABfrag af[4];
#pragma unroll
    for (int im = 0; im < 4; ++im) {
      const int row = wm + im * 16 + lr;
#pragma unroll
      for (int i = 0; i < 8; ++i) {
        const int k = 16 * hi + 2 * i;
        af[im].u[i] = *(const uint32_t*)&sA[buf][row][k];
      }
    }
    // B fragments from transposed tile: lane = col, K pairs 2j/2j+1 (+16 hi)
    ABfrag bfr[2];
#pragma unroll
    for (int jn = 0; jn < 2; ++jn) {
      const int col = wn + jn * 16 + lr;
#pragma unroll
      for (int j = 0; j < 8; ++j)
        bfr[jn].u[j] = *(const uint32_t*)&sBt[buf][col][16 * hi + 2 * j];
    }
#pragma unroll
    for (int im = 0; im < 4; ++im)
#pragma unroll
      for (int jn = 0; jn < 2; ++jn)
        acc[im][jn] = __builtin_amdgcn_wmma_f32_16x16x32_bf16(
            false, af[im].v, false, bfr[jn].v, (short)0, acc[im][jn], false, false);

    if (more) commitB(buf ^ 1);          // reg->LDS after compute
    wait_async();
    __syncthreads();
  }

  // C layout: VGPR r -> row r (lanes 0-15) / row 8+r (lanes 16-31), col = lr
#pragma unroll
  for (int im = 0; im < 4; ++im)
#pragma unroll
    for (int jn = 0; jn < 2; ++jn)
#pragma unroll
      for (int r = 0; r < 8; ++r) {
        const int row = m0 + wm + im * 16 + r + 8 * hi;
        const int col = n0 + wn + jn * 16 + lr;
        storeC(&C[(size_t)row * N + col], acc[im][jn][r]);
      }
}

// ---------------------------------------------------------------------------
// RoPE + head packing: kv(b,s,3072)+q(b,s,2048) bf16 -> qf/k/v (b,h,s,128) bf16
// ---------------------------------------------------------------------------
__global__ void rope_pack_kernel(const uint16_t* __restrict__ kvb,
                                 const uint16_t* __restrict__ qb,
                                 const float* __restrict__ cosp,
                                 const float* __restrict__ sinp,
                                 uint16_t* __restrict__ qo,
                                 uint16_t* __restrict__ ko,
                                 uint16_t* __restrict__ vo) {
  const int t = blockIdx.x * blockDim.x + threadIdx.x;
  const int total = BB * SS * HH * 32;
  if (t >= total) return;
  const int i = t & 31;
  const int h = (t >> 5) & (HH - 1);
  const int s = (t >> 9) & (SS - 1);
  const int b = t / (32 * HH * SS);

  const size_t tok = (size_t)b * SS + s;
  const uint16_t* kvrow = kvb + tok * 3072;   // [k(1024) | sh(1024) | v(1024)]
  const uint16_t* qrow  = qb + tok * 2048;    // [q1(1024) | q2(1024)]
  const float c  = cosp[s * 32 + i];
  const float sn = sinp[s * 32 + i];
  const size_t ob = ((size_t)(b * HH + h) * SS + s) * 128;
  const int e = 2 * i;

  // qf = [q1 | rope(q2)]
  qo[ob + e]     = qrow[h * 64 + e];
  qo[ob + e + 1] = qrow[h * 64 + e + 1];
  {
    const float xr = bf2f(qrow[1024 + h * 64 + e]);
    const float xi = bf2f(qrow[1024 + h * 64 + e + 1]);
    qo[ob + 64 + e]     = f2bf(xr * c - xi * sn);
    qo[ob + 64 + e + 1] = f2bf(xr * sn + xi * c);
  }
  // k = [sh | rope(k)] ; v = [sh | v]
  const uint16_t sh0 = kvrow[1024 + h * 64 + e];
  const uint16_t sh1 = kvrow[1024 + h * 64 + e + 1];
  ko[ob + e] = sh0; ko[ob + e + 1] = sh1;
  {
    const float kr = bf2f(kvrow[h * 64 + e]);
    const float ki = bf2f(kvrow[h * 64 + e + 1]);
    ko[ob + 64 + e]     = f2bf(kr * c - ki * sn);
    ko[ob + 64 + e + 1] = f2bf(kr * sn + ki * c);
  }
  vo[ob + e] = sh0; vo[ob + e + 1] = sh1;
  vo[ob + 64 + e]     = kvrow[2048 + h * 64 + e];
  vo[ob + 64 + e + 1] = kvrow[2048 + h * 64 + e + 1];
}

// ---------------------------------------------------------------------------
// Flash attention (causal), d=128, bf16 WMMA, online softmax.
// Block = 128 threads (4 waves); each wave owns 16 q-rows; kv tiles of 32.
// Double-buffered K (async DMA) and V (fetch/commit) LDS tiles so the
// next tile streams in behind the current tile's 16 WMMAs.
// ---------------------------------------------------------------------------
__global__ __launch_bounds__(128) void attn_fa_kernel(const uint16_t* __restrict__ Q,
                                                      const uint16_t* __restrict__ K,
                                                      const uint16_t* __restrict__ V,
                                                      uint16_t* __restrict__ O) {
  constexpr int LDK = 136;  // 272B rows keep 16B async chunks aligned
  constexpr int LDV = 34;   // V^T tile
  constexpr int LDP = 34;   // per-wave P staging
  __shared__ uint16_t sK[2][32][LDK];
  __shared__ uint16_t sVt[2][128][LDV];
  __shared__ uint16_t sP[4][16][LDP];

  const int tid  = threadIdx.x;
  const int lane = tid & 31, wid = tid >> 5;
  const int hi = lane >> 4, lr = lane & 15;
  const int bh = blockIdx.y;
  const int q0 = blockIdx.x * 64;
  const int qrow0 = q0 + wid * 16;

  // Q A-fragments: 16 rows x 128 d as four 16x32 chunks (held for whole loop)
  ABfrag qf[4];
  {
    const uint16_t* qptr = Q + ((size_t)bh * SS + (qrow0 + lr)) * 128;
#pragma unroll
    for (int t = 0; t < 4; ++t)
#pragma unroll
      for (int i = 0; i < 8; ++i) {
        const int k = 32 * t + 16 * hi + 2 * i;
        qf[t].u[i] = *(const uint32_t*)(qptr + k);
      }
  }

  auto asyncK = [&](int buf, int kv0) {
#pragma unroll
    for (int j = 0; j < 4; ++j) {
      const int c = tid + j * 128;      // 512 chunks: 32 rows x 16
      const int row = c >> 4, cc = (c & 15) * 8;
      async_copy_b128(K + ((size_t)bh * SS + kv0 + row) * 128 + cc,
                      &sK[buf][row][cc]);
    }
  };
  Q16 vreg[4];
  auto fetchV = [&](int kv0) {
#pragma unroll
    for (int j = 0; j < 4; ++j) {
      const int c = tid + j * 128;
      const int row = c >> 4, cc = (c & 15) * 8;
      vreg[j].q = *(const uint4*)(V + ((size_t)bh * SS + kv0 + row) * 128 + cc);
    }
  };
  auto commitV = [&](int buf) {
#pragma unroll
    for (int j = 0; j < 4; ++j) {
      const int c = tid + j * 128;
      const int row = c >> 4, cc = (c & 15) * 8;
#pragma unroll
      for (int e = 0; e < 8; ++e) sVt[buf][cc + e][row] = vreg[j].h[e];
    }
  };

  float m8[8], l8[8], alpha[8];
  v8f oacc[8];
#pragma unroll
  for (int r = 0; r < 8; ++r) { m8[r] = -1e30f; l8[r] = 0.0f; }
#pragma unroll
  for (int t = 0; t < 8; ++t) oacc[t] = vzero();

  const int nk = q0 / 32 + 2;  // causal: kv < q0+64
  fetchV(0);
  asyncK(0, 0);
  commitV(0);
  wait_async();
  __syncthreads();

  for (int kt = 0; kt < nk; ++kt) {
    const int kv0 = kt * 32;
    const int buf = kt & 1;
    const bool more = (kt + 1) < nk;
    if (more) {
      asyncK(buf ^ 1, kv0 + 32);
      fetchV(kv0 + 32);
      if (kt + 2 < nk) {
        const size_t pb = ((size_t)bh * SS + kv0 + 64 + (tid >> 2)) * 128;
        __builtin_prefetch(K + pb, 0, 0);
        __builtin_prefetch(V + pb, 0, 0);
      }
    }

    // S = Q K^T : two 16x16 tiles, accumulate over d in 4 WMMA steps each
    v8f sc[2];
#pragma unroll
    for (int jn = 0; jn < 2; ++jn) {
      v8f a = vzero();
#pragma unroll
      for (int t = 0; t < 4; ++t) {
        ABfrag kb;
#pragma unroll
        for (int j = 0; j < 8; ++j)
          kb.u[j] = *(const uint32_t*)&sK[buf][jn * 16 + lr][32 * t + 16 * hi + 2 * j];
        a = __builtin_amdgcn_wmma_f32_16x16x32_bf16(
            false, qf[t].v, false, kb.v, (short)0, a, false, false);
      }
      sc[jn] = a;
    }

    // Online softmax (row = r + 8*hi lives in one 16-lane half -> xor reduce)
    const float scale = 0.08838834764831845f;  // 1/sqrt(128)
#pragma unroll
    for (int r = 0; r < 8; ++r) {
      const int row = qrow0 + r + 8 * hi;
      float v0 = sc[0][r] * scale;
      float v1 = sc[1][r] * scale;
      v0 = (kv0 + lr <= row) ? v0 : -1e30f;
      v1 = (kv0 + 16 + lr <= row) ? v1 : -1e30f;
      float mx = fmaxf(v0, v1);
      mx = fmaxf(mx, __shfl_xor(mx, 1));
      mx = fmaxf(mx, __shfl_xor(mx, 2));
      mx = fmaxf(mx, __shfl_xor(mx, 4));
      mx = fmaxf(mx, __shfl_xor(mx, 8));
      const float mnew = fmaxf(m8[r], mx);
      alpha[r] = __expf(m8[r] - mnew);
      m8[r] = mnew;
      const float p0 = __expf(v0 - mnew);
      const float p1 = __expf(v1 - mnew);
      float su = p0 + p1;
      su += __shfl_xor(su, 1);
      su += __shfl_xor(su, 2);
      su += __shfl_xor(su, 4);
      su += __shfl_xor(su, 8);
      l8[r] = l8[r] * alpha[r] + su;
      const int rr = r + 8 * hi;             // C-layout -> LDS staging for A
      sP[wid][rr][lr]      = f2bf(p0);
      sP[wid][rr][16 + lr] = f2bf(p1);
    }
#pragma unroll
    for (int t = 0; t < 8; ++t)
#pragma unroll
      for (int r = 0; r < 8; ++r) oacc[t][r] *= alpha[r];

    // O += P (16x32) x V (32x128)
    ABfrag pf;
#pragma unroll
    for (int i = 0; i < 8; ++i) {
      const int k = 16 * hi + 2 * i;
      pf.u[i] = *(const uint32_t*)&sP[wid][lr][k];
    }
#pragma unroll
    for (int t = 0; t < 8; ++t) {
      ABfrag vf;
#pragma unroll
      for (int j = 0; j < 8; ++j)
        vf.u[j] = *(const uint32_t*)&sVt[buf][t * 16 + lr][16 * hi + 2 * j];
      oacc[t] = __builtin_amdgcn_wmma_f32_16x16x32_bf16(
          false, pf.v, false, vf.v, (short)0, oacc[t], false, false);
    }

    if (more) commitV(buf ^ 1);          // reg->LDS after compute
    wait_async();
    __syncthreads();
  }

  // Normalize and write attention output in token-major (b, s, h*128+d) bf16
  const int b = bh >> 4, h = bh & 15;
#pragma unroll
  for (int t = 0; t < 8; ++t)
#pragma unroll
    for (int r = 0; r < 8; ++r) {
      const int row = qrow0 + r + 8 * hi;
      const float val = oacc[t][r] / l8[r];
      O[((size_t)b * SS + row) * DD + h * 128 + t * 16 + lr] = f2bf(val);
    }
}

// ---------------------------------------------------------------------------
// Launch: convert -> kv/q GEMMs -> rope/pack -> flash attention -> proj GEMM
// Workspace use: ~139 MB of d_ws.
// ---------------------------------------------------------------------------
extern "C" void kernel_launch(void* const* d_in, const int* in_sizes, int n_in,
                              void* d_out, int out_size, void* d_ws, size_t ws_size,
                              hipStream_t stream) {
  const float* x     = (const float*)d_in[0];   // (B,S,D)
  const float* wkv   = (const float*)d_in[1];   // (D, 3D/2)
  const float* wq    = (const float*)d_in[2];   // (D, D)
  const float* wproj = (const float*)d_in[3];   // (D, D)
  const float* fcos  = (const float*)d_in[4];   // (S, 32)
  const float* fsin  = (const float*)d_in[5];   // (S, 32)
  (void)in_sizes; (void)n_in; (void)out_size; (void)ws_size;

  const int TOK = BB * SS;           // 4096
  char* ws = (char*)d_ws;
  size_t off = 0;
  auto alloc = [&](size_t bytes) -> void* {
    void* p = ws + off;
    off = (off + bytes + 255) & ~(size_t)255;
    return p;
  };
  uint16_t* xb     = (uint16_t*)alloc((size_t)TOK * DD * 2);        // 16 MB
  uint16_t* wkvb   = (uint16_t*)alloc((size_t)DD * 3072 * 2);       // 12 MB
  uint16_t* wqb    = (uint16_t*)alloc((size_t)DD * DD * 2);         // 8 MB
  uint16_t* wprojb = (uint16_t*)alloc((size_t)DD * DD * 2);         // 8 MB
  uint16_t* kvbf   = (uint16_t*)alloc((size_t)TOK * 3072 * 2);      // 24 MB
  uint16_t* qlin   = (uint16_t*)alloc((size_t)TOK * DD * 2);        // 16 MB
  uint16_t* qo     = (uint16_t*)alloc((size_t)BB * HH * SS * 128 * 2);
  uint16_t* ko     = (uint16_t*)alloc((size_t)BB * HH * SS * 128 * 2);
  uint16_t* vo     = (uint16_t*)alloc((size_t)BB * HH * SS * 128 * 2);
  uint16_t* attnbf = xb;  // x no longer needed once both GEMMs are done

  const int nx = TOK * DD;           // 8388608
  const int nwkv = DD * 3072;        // 6291456
  const int nwq = DD * DD;           // 4194304
  f32_to_bf16_kernel<<<(nx + 255) / 256, 256, 0, stream>>>(x, xb, nx);
  f32_to_bf16_kernel<<<(nwkv + 255) / 256, 256, 0, stream>>>(wkv, wkvb, nwkv);
  f32_to_bf16_kernel<<<(nwq + 255) / 256, 256, 0, stream>>>(wq, wqb, nwq);
  f32_to_bf16_kernel<<<(nwq + 255) / 256, 256, 0, stream>>>(wproj, wprojb, nwq);

  gemm_bf16<uint16_t><<<dim3(3072 / 128, TOK / 128), 256, 0, stream>>>(
      xb, wkvb, kvbf, TOK, 3072, DD);
  gemm_bf16<uint16_t><<<dim3(DD / 128, TOK / 128), 256, 0, stream>>>(
      xb, wqb, qlin, TOK, DD, DD);

  const int nrope = BB * SS * HH * 32;  // 2097152
  rope_pack_kernel<<<(nrope + 255) / 256, 256, 0, stream>>>(
      kvbf, qlin, fcos, fsin, qo, ko, vo);

  attn_fa_kernel<<<dim3(SS / 64, BB * HH), 128, 0, stream>>>(qo, ko, vo, attnbf);

  gemm_bf16<float><<<dim3(DD / 128, TOK / 128), 256, 0, stream>>>(
      attnbf, wprojb, (float*)d_out, TOK, DD, DD);
}
